// MultiHeadAttention_55671366090988
// MI455X (gfx1250) — matrix operand order repaired and checked by reading, not compiled, not run on hardware
//
#include <hip/hip_runtime.h>
#include <hip/hip_bf16.h>

// ---------------------------------------------------------------------------
// MHA with KV cache for MI455X (gfx1250): bf16 WMMA GEMMs + flash attention
// with double-buffered async global->LDS K/V staging (ASYNCcnt pipeline).
//   out[0 .. B*S*D)                       : output (f32)
//   out[B*S*D .. +B*2*H*(P+S)*64)         : present (f32)
// Workspace (48 MB):
//   q_bf  [B,H,S,64]      bf16 @ 0
//   k_bf  [B,H,P+S,64]    bf16 @ 8 MiB
//   v_bf  [B,H,P+S,64]    bf16 @ 24 MiB
//   merged[B,S,D]         bf16 @ 40 MiB
// ---------------------------------------------------------------------------

typedef __attribute__((ext_vector_type(16))) __bf16       v16bf;
typedef __attribute__((ext_vector_type(8)))  float        v8f;
typedef __attribute__((ext_vector_type(4)))  unsigned int u32x4;
typedef __attribute__((ext_vector_type(4)))  int          i32x4;

#define AS1 __attribute__((address_space(1)))
#define AS3 __attribute__((address_space(3)))

#define B_     2
#define S_     2048
#define D_     1024
#define H_     16
#define P_     2048
#define PS_    4096
#define N3_    3072

#if __has_builtin(__builtin_amdgcn_global_load_async_to_lds_b128)
#define USE_ASYNC 1
#else
#define USE_ASYNC 0
#endif

union FragU  { v16bf v; u32x4 u[2]; };
union U128bf { u32x4 u; __bf16 h[8]; };

// 16-byte global -> LDS copy: async (ASYNCcnt) when available, sync fallback.
__device__ inline void cp16_g2l(const void* gsrc, void* ldst) {
#if USE_ASYNC
  __builtin_amdgcn_global_load_async_to_lds_b128(
      (AS1 i32x4*)(void*)gsrc, (AS3 i32x4*)ldst, 0, 0);
#else
  *(u32x4*)ldst = *(const u32x4*)gsrc;
#endif
}

__device__ inline void wait_async0() {
#if USE_ASYNC
#if __has_builtin(__builtin_amdgcn_s_wait_asynccnt)
  __builtin_amdgcn_s_wait_asynccnt(0);
#else
  asm volatile("s_wait_asynccnt 0" ::: "memory");
#endif
#endif
}

// A fragment, 16x32 bf16, from row-major [16][ld] storage.
// lane<16: row=lane, K 0..15; lane>=16: K 16..31
__device__ inline v16bf load_frag_a(const __bf16* base, int ld) {
  int lane = threadIdx.x & 31;
  int row  = lane & 15;
  int kh   = (lane >> 4) & 1;
  const __bf16* p0 = base + row * ld + kh * 16;
  FragU f;
  f.u[0] = *(const u32x4*)(p0);
  f.u[1] = *(const u32x4*)(p0 + 8);
  return f.v;
}

// B fragment, K=32 x N=16, from B^T row-major [N][ld] storage (contig in K).
// lane n=lane&15 holds K (lane>>4)*16 .. +15
__device__ inline v16bf load_frag_b(const __bf16* base, int ld) {
  int lane = threadIdx.x & 31;
  int n    = lane & 15;
  int koff = (lane >> 4) << 4;
  const __bf16* p = base + n * ld + koff;
  FragU f;
  f.u[0] = *(const u32x4*)(p);
  f.u[1] = *(const u32x4*)(p + 8);
  return f.v;
}

__device__ inline v8f wmma_bf16(v16bf a, v16bf b, v8f c) {
  return __builtin_amdgcn_wmma_f32_16x16x32_bf16(false, a, false, b,
                                                 (short)0, c, false, false);
}

// ---------------------------------------------------------------------------
// Kernel 1: copy past KV into present output (f32) and bf16 mirrors.
// past [B,2,H,P,64] -> present [B,2,H,P+S,64] rows [0,P), k_bf/v_bf [B,H,P+S,64]
// ---------------------------------------------------------------------------
__global__ __launch_bounds__(256) void copy_past_kernel(
    const float* __restrict__ past, float* __restrict__ present,
    __bf16* __restrict__ k_bf, __bf16* __restrict__ v_bf) {
  size_t i = (size_t)blockIdx.x * 256 + threadIdx.x;   // < B*2*H*P*64 = 2^23
  int dep  = (int)(i & 63);
  int p    = (int)((i >> 6) & 2047);
  int rest = (int)(i >> 17);          // (b*2+kv)*16 + h
  int h    = rest & 15;
  int bkv  = rest >> 4;
  int kv   = bkv & 1;
  int b    = bkv >> 1;
  float val = past[i];
  present[((size_t)rest * PS_ + p) * 64 + dep] = val;
  size_t bfidx = (((size_t)(b * H_ + h) * PS_) + p) * 64 + dep;
  if (kv == 0) k_bf[bfidx] = (__bf16)val;
  else         v_bf[bfidx] = (__bf16)val;
}

// ---------------------------------------------------------------------------
// Kernel 2: qkv = x @ w_attn + b_attn  (M=4096, N=3072, K=1024), bf16 WMMA.
// WG tile 128x64, 8 waves of 32x32, K-step 32. Epilogue scatters q/k/v.
// ---------------------------------------------------------------------------
__global__ __launch_bounds__(256) void qkv_gemm_kernel(
    const float* __restrict__ x, const float* __restrict__ w_attn,
    const float* __restrict__ b_attn,
    __bf16* __restrict__ q_bf, __bf16* __restrict__ k_bf,
    __bf16* __restrict__ v_bf, float* __restrict__ present) {
  __shared__ __align__(16) __bf16 sA[128][32];
  __shared__ __align__(16) __bf16 sBt[64][32];   // B^T: [n][k]
  const int t    = threadIdx.x;
  const int lane = t & 31;
  const int wave = t >> 5;
  const int wm   = wave & 3;   // 4 waves along M
  const int wn   = wave >> 2;  // 2 waves along N
  const int n0   = blockIdx.x * 64;
  const int m0   = blockIdx.y * 128;

  v8f acc[2][2] = {};
  for (int k0 = 0; k0 < D_; k0 += 32) {
    for (int e = t; e < 128 * 32; e += 256) {
      int m = e >> 5, k = e & 31;
      sA[m][k] = (__bf16)x[(size_t)(m0 + m) * D_ + k0 + k];
    }
    for (int e = t; e < 64 * 32; e += 256) {
      int n = e & 63, k = e >> 6;
      sBt[n][k] = (__bf16)w_attn[(size_t)(k0 + k) * N3_ + n0 + n];
    }
    __syncthreads();
    v16bf a0 = load_frag_a(&sA[wm * 32][0], 32);
    v16bf a1 = load_frag_a(&sA[wm * 32 + 16][0], 32);
    v16bf b0 = load_frag_b(&sBt[wn * 32][0], 32);
    v16bf b1 = load_frag_b(&sBt[wn * 32 + 16][0], 32);
    acc[0][0] = wmma_bf16(a0, b0, acc[0][0]);
    acc[0][1] = wmma_bf16(a0, b1, acc[0][1]);
    acc[1][0] = wmma_bf16(a1, b0, acc[1][0]);
    acc[1][1] = wmma_bf16(a1, b1, acc[1][1]);
    __syncthreads();
  }

  const int hi = lane >> 4, col = lane & 15;
  for (int im = 0; im < 2; ++im)
    for (int in = 0; in < 2; ++in)
#pragma unroll
      for (int i = 0; i < 8; ++i) {
        int m = m0 + wm * 32 + im * 16 + i + 8 * hi;
        int n = n0 + wn * 32 + in * 16 + col;
        float val = acc[im][in][i] + b_attn[n];
        int b = m >> 11, s = m & 2047;
        int sec = n >> 10, d_idx = n & 1023;
        int h = d_idx >> 6, dep = d_idx & 63;
        if (sec == 0) {
          q_bf[(((size_t)(b * H_ + h) * S_) + s) * 64 + dep] = (__bf16)val;
        } else {
          int kv = sec - 1;
          present[((((size_t)(b * 2 + kv) * H_ + h) * PS_) + P_ + s) * 64 + dep] = val;
          size_t bfidx = (((size_t)(b * H_ + h) * PS_) + P_ + s) * 64 + dep;
          if (kv == 0) k_bf[bfidx] = (__bf16)val;
          else         v_bf[bfidx] = (__bf16)val;
        }
      }
}

// ---------------------------------------------------------------------------
// Kernel 3: flash attention. WG = 128 q rows (8 waves x 16). Streams 32-key
// chunks with double-buffered async global->LDS loads (ASYNCcnt pipeline):
//   wait(chunk c) -> barrier -> prefetch(c+1) -> V transpose -> barrier ->
//   QK^T / online softmax / PV via bf16 WMMA.
// ---------------------------------------------------------------------------
__global__ __launch_bounds__(256) void flash_attn_kernel(
    const __bf16* __restrict__ q_bf, const __bf16* __restrict__ k_bf,
    const __bf16* __restrict__ v_bf, __bf16* __restrict__ merged) {
  __shared__ __align__(16) __bf16 sQ[8][16][64];      // per-wave Q tile (16K)
  __shared__ __align__(16) __bf16 sK[2][32][64];      // keys [key][d]     (8K)
  __shared__ __align__(16) __bf16 sVraw[2][32][64];   // V raw [key][d]    (8K)
  __shared__ __align__(16) __bf16 sVt[64][32];        // V^T [d][key]      (4K)
  __shared__ __align__(16) __bf16 sP[8][16][32];      // per-wave P        (8K)
  const int t    = threadIdx.x;
  const int lane = t & 31;
  const int w    = t >> 5;
  const int q0   = blockIdx.x * 128;
  const int hh   = blockIdx.y;
  const int bb   = blockIdx.z;
  const int hi   = lane >> 4, col = lane & 15;
  const size_t kvbase = (size_t)(bb * H_ + hh) * PS_ * 64;

  // stage this wave's 16x64 Q tile (contiguous in global)
  {
    const u32x4* src =
        (const u32x4*)(q_bf + ((size_t)(bb * H_ + hh) * S_ + q0 + w * 16) * 64);
    u32x4* dst = (u32x4*)(&sQ[w][0][0]);
    for (int r = lane; r < 128; r += 32) dst[r] = src[r];
  }
  asm volatile("s_wait_dscnt 0" ::: "memory");
  v16bf qa0 = load_frag_a(&sQ[w][0][0], 64);
  v16bf qa1 = load_frag_a(&sQ[w][0][32], 64);

  v8f o[4] = {};
  float m_[8], l_[8], alpha_[8];
#pragma unroll
  for (int i = 0; i < 8; ++i) { m_[i] = -1e30f; l_[i] = 0.f; }

  int kend = P_ + q0 + 128;
  if (kend > PS_) kend = PS_;
  const int   nch   = kend / 32;
  const float SCALE = 0.125f;   // 1/sqrt(64)

  // each thread stages 16B of K and 16B of V per chunk (256 thr * 16B = 4KB)
  auto issue_chunk = [&](int kt, int buf) {
    cp16_g2l(k_bf + kvbase + (size_t)kt * 64 + t * 8,
             (__bf16*)(&sK[buf][0][0]) + t * 8);
    cp16_g2l(v_bf + kvbase + (size_t)kt * 64 + t * 8,
             (__bf16*)(&sVraw[buf][0][0]) + t * 8);
  };

  issue_chunk(0, 0);
  for (int c = 0; c < nch; ++c) {
    const int buf = c & 1;
    const int kt  = c * 32;

    wait_async0();         // chunk c landed (issued one iteration ahead)
    __syncthreads();       // all waves done with previous compute + waits

    if (c + 1 < nch) issue_chunk(kt + 32, buf ^ 1);  // prefetch next chunk

    // transpose V chunk: sVraw[buf][key][d] -> sVt[d][key]
    {
      int key = t >> 3;
      int d8  = (t & 7) * 8;
      U128bf vv;
      vv.u = *(const u32x4*)(&sVraw[buf][key][d8]);
#pragma unroll
      for (int j = 0; j < 8; ++j) sVt[d8 + j][key] = vv.h[j];
    }
    __syncthreads();

    // S = Q K^T  (two 16x16 score tiles over 32 keys)
    v8f s0 = {}, s1 = {};
    v16bf b00 = load_frag_b(&sK[buf][0][0], 64);
    v16bf b01 = load_frag_b(&sK[buf][0][32], 64);
    v16bf b10 = load_frag_b(&sK[buf][16][0], 64);
    v16bf b11 = load_frag_b(&sK[buf][16][32], 64);
    s0 = wmma_bf16(qa0, b00, s0);
    s0 = wmma_bf16(qa1, b01, s0);
    s1 = wmma_bf16(qa0, b10, s1);
    s1 = wmma_bf16(qa1, b11, s1);

    float p0v[8], p1v[8];
#pragma unroll
    for (int i = 0; i < 8; ++i) {
      int row_local = i + 8 * hi;
      int qpos = P_ + q0 + w * 16 + row_local;
      float v0 = s0[i] * SCALE; if (kt + col > qpos)      v0 += -1e9f;
      float v1 = s1[i] * SCALE; if (kt + 16 + col > qpos) v1 += -1e9f;
      float rmax = fmaxf(v0, v1);
      rmax = fmaxf(rmax, __shfl_xor(rmax, 1));
      rmax = fmaxf(rmax, __shfl_xor(rmax, 2));
      rmax = fmaxf(rmax, __shfl_xor(rmax, 4));
      rmax = fmaxf(rmax, __shfl_xor(rmax, 8));
      float mnew = fmaxf(m_[i], rmax);
      float al = __expf(m_[i] - mnew);
      float p0 = __expf(v0 - mnew);
      float p1 = __expf(v1 - mnew);
      float rsum = p0 + p1;
      rsum += __shfl_xor(rsum, 1);
      rsum += __shfl_xor(rsum, 2);
      rsum += __shfl_xor(rsum, 4);
      rsum += __shfl_xor(rsum, 8);
      l_[i] = l_[i] * al + rsum;
      m_[i] = mnew;
      alpha_[i] = al;
      p0v[i] = p0; p1v[i] = p1;
    }
#pragma unroll
    for (int d4 = 0; d4 < 4; ++d4)
#pragma unroll
      for (int i = 0; i < 8; ++i) o[d4][i] *= alpha_[i];

    // stage P (C-layout -> A-layout via wave-private LDS)
#pragma unroll
    for (int i = 0; i < 8; ++i) {
      int row_local = i + 8 * hi;
      sP[w][row_local][col]      = (__bf16)p0v[i];
      sP[w][row_local][16 + col] = (__bf16)p1v[i];
    }
    asm volatile("s_wait_dscnt 0" ::: "memory");
    v16bf pa = load_frag_a(&sP[w][0][0], 32);
#pragma unroll
    for (int d4 = 0; d4 < 4; ++d4) {
      v16bf bv = load_frag_b(&sVt[d4 * 16][0], 32);
      o[d4] = wmma_bf16(pa, bv, o[d4]);
    }
  }

  // epilogue: normalize and write merged [B,S,D] (D index = h*64 + dep)
#pragma unroll
  for (int d4 = 0; d4 < 4; ++d4)
#pragma unroll
    for (int i = 0; i < 8; ++i) {
      int row_local = i + 8 * hi;
      int q = q0 + w * 16 + row_local;
      int dep = d4 * 16 + col;
      float val = o[d4][i] / l_[i];
      merged[((size_t)(bb * S_ + q)) * D_ + hh * 64 + dep] = (__bf16)val;
    }
}

// ---------------------------------------------------------------------------
// Kernel 4: output = merged @ w_proj + b_proj  (M=4096, N=1024, K=1024)
// ---------------------------------------------------------------------------
__global__ __launch_bounds__(256) void proj_gemm_kernel(
    const __bf16* __restrict__ merged, const float* __restrict__ w_proj,
    const float* __restrict__ b_proj, float* __restrict__ out) {
  __shared__ __align__(16) __bf16 sA[128][32];
  __shared__ __align__(16) __bf16 sBt[64][32];
  const int t    = threadIdx.x;
  const int lane = t & 31;
  const int wave = t >> 5;
  const int wm   = wave & 3;
  const int wn   = wave >> 2;
  const int n0   = blockIdx.x * 64;
  const int m0   = blockIdx.y * 128;

  v8f acc[2][2] = {};
  for (int k0 = 0; k0 < D_; k0 += 32) {
    for (int c = t; c < 512; c += 256) {        // 128x32 bf16, 8-elt vectors
      int m = c >> 2;
      int k = (c & 3) * 8;
      *(u32x4*)(&sA[m][k]) =
          *(const u32x4*)(merged + (size_t)(m0 + m) * D_ + k0 + k);
    }
    for (int e = t; e < 2048; e += 256) {
      int n = e & 63, k = e >> 6;
      sBt[n][k] = (__bf16)w_proj[(size_t)(k0 + k) * D_ + n0 + n];
    }
    __syncthreads();
    v16bf a0 = load_frag_a(&sA[wm * 32][0], 32);
    v16bf a1 = load_frag_a(&sA[wm * 32 + 16][0], 32);
    v16bf b0 = load_frag_b(&sBt[wn * 32][0], 32);
    v16bf b1 = load_frag_b(&sBt[wn * 32 + 16][0], 32);
    acc[0][0] = wmma_bf16(a0, b0, acc[0][0]);
    acc[0][1] = wmma_bf16(a0, b1, acc[0][1]);
    acc[1][0] = wmma_bf16(a1, b0, acc[1][0]);
    acc[1][1] = wmma_bf16(a1, b1, acc[1][1]);
    __syncthreads();
  }

  const int hi = lane >> 4, col = lane & 15;
  for (int im = 0; im < 2; ++im)
    for (int in = 0; in < 2; ++in)
#pragma unroll
      for (int i = 0; i < 8; ++i) {
        int m = m0 + wm * 32 + im * 16 + i + 8 * hi;
        int n = n0 + wn * 32 + in * 16 + col;
        out[(size_t)m * D_ + n] = acc[im][in][i] + b_proj[n];
      }
}

// ---------------------------------------------------------------------------
extern "C" void kernel_launch(void* const* d_in, const int* in_sizes, int n_in,
                              void* d_out, int out_size, void* d_ws, size_t ws_size,
                              hipStream_t stream) {
  const float* x      = (const float*)d_in[0];
  // d_in[1] = mask: unused, the causal+cache mask is computed analytically.
  const float* past   = (const float*)d_in[2];
  const float* w_attn = (const float*)d_in[3];
  const float* b_attn = (const float*)d_in[4];
  const float* w_proj = (const float*)d_in[5];
  const float* b_proj = (const float*)d_in[6];

  float* out     = (float*)d_out;
  float* present = out + (size_t)B_ * S_ * D_;

  char* ws = (char*)d_ws;
  __bf16* q_bf   = (__bf16*)(ws);                              //  8 MiB
  __bf16* k_bf   = (__bf16*)(ws + 8388608);                    // 16 MiB
  __bf16* v_bf   = (__bf16*)(ws + 8388608 + 16777216);         // 16 MiB
  __bf16* merged = (__bf16*)(ws + 8388608 + 2 * 16777216);     //  8 MiB

  copy_past_kernel<<<32768, 256, 0, stream>>>(past, present, k_bf, v_bf);
  qkv_gemm_kernel<<<dim3(N3_ / 64, (B_ * S_) / 128), 256, 0, stream>>>(
      x, w_attn, b_attn, q_bf, k_bf, v_bf, present);
  flash_attn_kernel<<<dim3(S_ / 128, H_, B_), 256, 0, stream>>>(
      q_bf, k_bf, v_bf, merged);
  proj_gemm_kernel<<<dim3(D_ / 64, (B_ * S_) / 128), 256, 0, stream>>>(
      merged, w_proj, b_proj, out);
}
